// Net_86852828660168
// MI455X (gfx1250) — matrix operand order repaired and checked by reading, not compiled, not run on hardware
//
#include <hip/hip_runtime.h>
#include <hip/hip_bf16.h>

// ---------------------------------------------------------------------------
// Fused soft-pool + dual-branch 5-layer MLP + cross-mul for MI455X (gfx1250).
// Wave32; each wave owns 16 batch rows; workgroup = 8 waves = 128 rows.
// Layer1 weights staged as bf16 in LDS (uses the 320KB WGP LDS).
// All GEMMs via v_wmma_f32_16x16x32_bf16 with f32 accumulation.
// ---------------------------------------------------------------------------

typedef float v8f  __attribute__((ext_vector_type(8)));
typedef __bf16 v16bf __attribute__((ext_vector_type(16)));

#define D_IN     768
#define W1ROWS   96
#define WSTRIDE  776   // 768 + 8 bf16 pad (bank stagger)
#define ASTRIDE  104   // 96 + 8 bf16 pad, per-wave activation scratch stride
#define NWAVES   8
#define SMEM_ELEMS (2 * W1ROWS * WSTRIDE + NWAVES * 16 * ASTRIDE)
#define SMEM_BYTES (SMEM_ELEMS * 2)

static __device__ __forceinline__ v8f wmma_bf16(v16bf a, v16bf b, v8f c) {
  return __builtin_amdgcn_wmma_f32_16x16x32_bf16(false, a, false, b, (short)0, c,
                                                 false, false);
}

// B fragment (32x16, K-major per ISA layout) from LDS-resident bf16 W1.
// b[i] = W1[tile*16 + n][kc + kh*8 + i + (i>=8 ? 8 : 0)], n = lane&15, kh = lane>>4.
static __device__ __forceinline__ v16bf ld_b_frag_lds(const __bf16* w1, int tile, int kc) {
  const int n  = threadIdx.x & 15;
  const int kh = (threadIdx.x >> 4) & 1;
  const __bf16* p = w1 + (size_t)(tile * 16 + n) * WSTRIDE + kc + kh * 8;
  union { v16bf v; uint4 q[2]; } f;
  f.q[0] = *(const uint4*)(p);       // ds_load_b128
  f.q[1] = *(const uint4*)(p + 16);  // ds_load_b128
  return f.v;
}

// B fragment from global fp32 weight [out_dim, in_dim], zero-padded to 16x32.
static __device__ __forceinline__ v16bf ld_b_frag_global(const float* __restrict__ W,
                                                         int out_dim, int in_dim,
                                                         int tile, int kc) {
  const int n   = threadIdx.x & 15;
  const int kh  = (threadIdx.x >> 4) & 1;
  const int row = tile * 16 + n;
  const int kb  = kc + kh * 8;
  union { v16bf v; __bf16 h[16]; } f;
#pragma unroll
  for (int i = 0; i < 16; ++i) {
    const int k = kb + i + (i >= 8 ? 8 : 0);
    float x = (row < out_dim && k < in_dim) ? W[(size_t)row * in_dim + k] : 0.0f;
    f.h[i] = (__bf16)x;
  }
  return f.v;
}

// A fragment (16x32) from per-wave LDS activation scratch (rows x ASTRIDE).
// a[i<8]  = act[m][kc + kh*8 + i]
// a[i>=8] = act[m][kc + 16 + kh*8 + (i-8)]   (matches ISA 16-bit A layout)
static __device__ __forceinline__ v16bf ld_a_frag_lds(const __bf16* act, int kc) {
  const int m  = threadIdx.x & 15;
  const int kh = (threadIdx.x >> 4) & 1;
  const __bf16* p = act + m * ASTRIDE + kc + kh * 8;
  union { v16bf v; uint4 q[2]; } f;
  f.q[0] = *(const uint4*)(p);        // run 0 (8 bf16)
  f.q[1] = *(const uint4*)(p + 16);   // run 1 (8 bf16)
  return f.v;
}

// Store a C tile (f32) as relu->bf16 into activation scratch.
static __device__ __forceinline__ void st_act_tile(__bf16* act, int t, v8f acc) {
  const int n  = threadIdx.x & 15;
  const int rh = (threadIdx.x >> 4) & 1;
#pragma unroll
  for (int r = 0; r < 8; ++r) {
    float v = fmaxf(acc[r], 0.0f);
    act[(r + rh * 8) * ASTRIDE + t * 16 + n] = (__bf16)v;
  }
}

static __device__ __forceinline__ void st_zero_tile(__bf16* act, int t) {
  const int n  = threadIdx.x & 15;
  const int rh = (threadIdx.x >> 4) & 1;
#pragma unroll
  for (int r = 0; r < 8; ++r)
    act[(r + rh * 8) * ASTRIDE + t * 16 + n] = (__bf16)0.0f;
}

#define LDS_FENCE() asm volatile("s_wait_dscnt 0" ::: "memory")

// Layers 2..5 for one branch: 96->48->24->12->out5 (no relu after last).
static __device__ v8f run_tail(const v8f acc1[6], __bf16* act,
                               const float* __restrict__ W2, const float* __restrict__ W3,
                               const float* __restrict__ W4, const float* __restrict__ W5,
                               int out5) {
  const v8f vz = {0.f, 0.f, 0.f, 0.f, 0.f, 0.f, 0.f, 0.f};

  // ---- layer 2: K=96 (6 act tiles), N=48 (3 tiles)
  LDS_FENCE();
#pragma unroll
  for (int t = 0; t < 6; ++t) st_act_tile(act, t, acc1[t]);
  LDS_FENCE();
  v8f a2[3] = {vz, vz, vz};
#pragma unroll
  for (int kc = 0; kc < 96; kc += 32) {
    v16bf a = ld_a_frag_lds(act, kc);
#pragma unroll
    for (int t = 0; t < 3; ++t)
      a2[t] = wmma_bf16(a, ld_b_frag_global(W2, 48, 96, t, kc), a2[t]);
  }

  // ---- layer 3: K=48 (pad to 64), N=24 (2 tiles; cols 24..31 auto-zero)
  LDS_FENCE();
#pragma unroll
  for (int t = 0; t < 3; ++t) st_act_tile(act, t, a2[t]);
  st_zero_tile(act, 3);
  LDS_FENCE();
  v8f a3[2] = {vz, vz};
#pragma unroll
  for (int kc = 0; kc < 64; kc += 32) {
    v16bf a = ld_a_frag_lds(act, kc);
#pragma unroll
    for (int t = 0; t < 2; ++t)
      a3[t] = wmma_bf16(a, ld_b_frag_global(W3, 24, 48, t, kc), a3[t]);
  }

  // ---- layer 4: K=24 (pad to 32), N=12 (1 tile; cols 12..15 auto-zero)
  LDS_FENCE();
  st_act_tile(act, 0, a3[0]);
  st_act_tile(act, 1, a3[1]);
  LDS_FENCE();
  v8f a4 = vz;
  {
    v16bf a = ld_a_frag_lds(act, 0);
    a4 = wmma_bf16(a, ld_b_frag_global(W4, 12, 24, 0, 0), a4);
  }

  // ---- layer 5: K=12 (pad to 32), N=out5 (1 tile), no trailing relu
  LDS_FENCE();
  st_act_tile(act, 0, a4);
  st_zero_tile(act, 1);
  LDS_FENCE();
  v8f a5 = vz;
  {
    v16bf a = ld_a_frag_lds(act, 0);
    a5 = wmma_bf16(a, ld_b_frag_global(W5, out5, 12, 0, 0), a5);
  }
  return a5;
}

// Layer 1: fused pool(+relu,+bf16) from global, B from LDS. NSL = active slots.
template <int NSL>
static __device__ __forceinline__ void layer1(const float* __restrict__ res,
                                              size_t rowbase, const int* slot,
                                              const float* wE, const float* wS,
                                              const __bf16* w1e, const __bf16* w1s,
                                              v8f accE[6], v8f accS[6]) {
  const int kh = (threadIdx.x >> 4) & 1;
  const float* base[NSL];
#pragma unroll
  for (int si = 0; si < NSL; ++si)
    base[si] = res + rowbase + (size_t)slot[si] * D_IN;

  for (int kc = 0; kc < D_IN; kc += 32) {
    v16bf aE, aS;
#pragma unroll
    for (int run = 0; run < 2; ++run) {
      const int kb = kc + kh * 8 + run * 16;
      float pe[8] = {0, 0, 0, 0, 0, 0, 0, 0};
      float ps[8] = {0, 0, 0, 0, 0, 0, 0, 0};
#pragma unroll
      for (int si = 0; si < NSL; ++si) {
        float q[8];
        *(float4*)(&q[0]) = *(const float4*)(base[si] + kb);      // global_load_b128
        *(float4*)(&q[4]) = *(const float4*)(base[si] + kb + 4);  // global_load_b128
        const float we = wE[si], ws = wS[si];
#pragma unroll
        for (int i = 0; i < 8; ++i) { pe[i] += we * q[i]; ps[i] += ws * q[i]; }
      }
#pragma unroll
      for (int i = 0; i < 8; ++i) {
        aE[run * 8 + i] = (__bf16)fmaxf(pe[i], 0.0f);
        aS[run * 8 + i] = (__bf16)fmaxf(ps[i], 0.0f);
      }
    }
#pragma unroll
    for (int t = 0; t < 6; ++t)
      accE[t] = wmma_bf16(aE, ld_b_frag_lds(w1e, t, kc), accE[t]);
#pragma unroll
    for (int t = 0; t < 6; ++t)
      accS[t] = wmma_bf16(aS, ld_b_frag_lds(w1s, t, kc), accS[t]);
  }
}

__global__ __launch_bounds__(256, 1) void fused_mlp_kernel(
    const float* __restrict__ res,
    const float* __restrict__ sW1, const float* __restrict__ sW2,
    const float* __restrict__ sW3, const float* __restrict__ sW4,
    const float* __restrict__ sW5,
    const float* __restrict__ eW1, const float* __restrict__ eW2,
    const float* __restrict__ eW3, const float* __restrict__ eW4,
    const float* __restrict__ eW5,
    const float* __restrict__ s_seq, const float* __restrict__ s_pair,
    const float* __restrict__ e_seq, const float* __restrict__ e_pair,
    const float* __restrict__ cross_w,
    float* __restrict__ out) {
  extern __shared__ char smem_raw[];
  __bf16* w1e = (__bf16*)smem_raw;
  __bf16* w1s = w1e + W1ROWS * WSTRIDE;
  const int wave = threadIdx.x >> 5;
  __bf16* act = w1s + W1ROWS * WSTRIDE + wave * (16 * ASTRIDE);

  // ---- stage W1 (both branches) fp32 -> bf16 into LDS, row-padded ----------
  {
    const int n4 = (W1ROWS * D_IN) / 4;  // float4 elements per matrix
    for (int i = threadIdx.x; i < 2 * n4; i += 256) {
      const bool se = (i >= n4);
      const int j = se ? (i - n4) : i;
      const float4 v = ((const float4*)(se ? sW1 : eW1))[j];
      const int r = j / (D_IN / 4);
      const int c = (j - r * (D_IN / 4)) * 4;
      union { __bf16 h[4]; uint2 u; } pk;
      pk.h[0] = (__bf16)v.x; pk.h[1] = (__bf16)v.y;
      pk.h[2] = (__bf16)v.z; pk.h[3] = (__bf16)v.w;
      *(uint2*)((se ? w1s : w1e) + (size_t)r * WSTRIDE + c) = pk.u;
    }
  }
  __syncthreads();

  const int lane = threadIdx.x & 31;
  const int m = lane & 15;
  const int kh = lane >> 4;
  const int R0 = blockIdx.x * (NWAVES * 16) + wave * 16;
  const size_t rowbase = (size_t)(R0 + m) * 4 * D_IN;

  const float cw0 = cross_w[0], cw1 = cross_w[1];
  float outAcc[8];
#pragma unroll
  for (int r = 0; r < 8; ++r) outAcc[r] = 0.0f;

  const v8f vz = {0.f, 0.f, 0.f, 0.f, 0.f, 0.f, 0.f, 0.f};

  for (int p = 0; p < 2; ++p) {
    v8f accE[6], accS[6];
#pragma unroll
    for (int t = 0; t < 6; ++t) { accE[t] = vz; accS[t] = vz; }

    if (p == 0) {
      const int slot[2] = {0, 2};
      const float wE[2] = {e_pair[0], e_pair[1]};
      const float wS[2] = {s_pair[0], s_pair[1]};
      layer1<2>(res, rowbase, slot, wE, wS, w1e, w1s, accE, accS);
    } else {
      const int slot[4] = {0, 1, 2, 3};
      const float wE[4] = {e_seq[0], e_seq[1], e_seq[2], e_seq[3]};
      const float wS[4] = {s_seq[0], s_seq[1], s_seq[2], s_seq[3]};
      layer1<4>(res, rowbase, slot, wE, wS, w1e, w1s, accE, accS);
    }

    const v8f e5 = run_tail(accE, act, eW2, eW3, eW4, eW5, 8);
    const v8f s5 = run_tail(accS, act, sW2, sW3, sW4, sW5, 2);

    const float cw = (p == 0) ? cw0 : cw1;
#pragma unroll
    for (int r = 0; r < 8; ++r) {
      float es = e5[r];  // cols 8..15 are exact zeros (weight-padded)
      es += __shfl_xor(es, 1);
      es += __shfl_xor(es, 2);
      es += __shfl_xor(es, 4);
      es += __shfl_xor(es, 8);
      outAcc[r] += cw * s5[r] * es;  // z[k,h] = s[k,h] * sum_j e[k,j]
    }
  }

  // ---- store [B,2] output: lanes with col<2 hold valid data ---------------
  const int n = lane & 15;
  if (n < 2) {
#pragma unroll
    for (int r = 0; r < 8; ++r)
      out[(size_t)(R0 + r + kh * 8) * 2 + n] = outAcc[r];
  }
}

extern "C" void kernel_launch(void* const* d_in, const int* in_sizes, int n_in,
                              void* d_out, int out_size, void* d_ws, size_t ws_size,
                              hipStream_t stream) {
  const float* res    = (const float*)d_in[0];
  const float* sW1    = (const float*)d_in[1];
  const float* sW2    = (const float*)d_in[2];
  const float* sW3    = (const float*)d_in[3];
  const float* sW4    = (const float*)d_in[4];
  const float* sW5    = (const float*)d_in[5];
  const float* eW1    = (const float*)d_in[6];
  const float* eW2    = (const float*)d_in[7];
  const float* eW3    = (const float*)d_in[8];
  const float* eW4    = (const float*)d_in[9];
  const float* eW5    = (const float*)d_in[10];
  const float* s_seq  = (const float*)d_in[11];
  const float* s_pair = (const float*)d_in[12];
  const float* e_seq  = (const float*)d_in[13];
  const float* e_pair = (const float*)d_in[14];
  const float* crossw = (const float*)d_in[15];
  float* out = (float*)d_out;

  const int B = in_sizes[0] / (4 * D_IN);  // 65536
  const int rows_per_wg = NWAVES * 16;     // 128

  hipFuncSetAttribute((const void*)fused_mlp_kernel,
                      hipFuncAttributeMaxDynamicSharedMemorySize, SMEM_BYTES);

  fused_mlp_kernel<<<dim3(B / rows_per_wg), dim3(256), SMEM_BYTES, stream>>>(
      res, sW1, sW2, sW3, sW4, sW5, eW1, eW2, eW3, eW4, eW5,
      s_seq, s_pair, e_seq, e_pair, crossw, out);
}
